// MaskedAttention_59107339928123
// MI455X (gfx1250) — matrix operand order repaired and checked by reading, not compiled, not run on hardware
//
#include <hip/hip_runtime.h>

#define BB 4
#define SS 2048
#define DD 1024

typedef _Float16 v16h __attribute__((ext_vector_type(16)));
typedef _Float16 h8   __attribute__((ext_vector_type(8)));
typedef float    v8f  __attribute__((ext_vector_type(8)));
typedef float    f4   __attribute__((ext_vector_type(4)));
typedef unsigned int u32x4 __attribute__((ext_vector_type(4)));
typedef int          i32x4 __attribute__((ext_vector_type(4)));
typedef int          i32x8 __attribute__((ext_vector_type(8)));

#if defined(__has_builtin)
#if __has_builtin(__builtin_amdgcn_tensor_load_to_lds) && \
    __has_builtin(__builtin_amdgcn_s_wait_tensorcnt)
#define HAVE_TDM 1
#endif
#endif
#ifndef HAVE_TDM
#define HAVE_TDM 0
#endif

// ---- WMMA operand builders -------------------------------------------------
// A-matrix 16x32 f16 layout (wave32): lane&15 = M (row); lane>>4 selects K half.
//   halves 0..7  = K in [half*8,  half*8+7]
//   halves 8..15 = K in [16+half*8, 16+half*8+7]
// B-matrix 32x16 f16 layout: lane&15 = N (col); lane>>4 selects K half.
//   halves 0..7  = K in [half*8,  half*8+7]
//   halves 8..15 = K in [16+half*8, 16+half*8+7]

__device__ __forceinline__ v16h cat_h8(h8 a, h8 b) {
  v16h r;
#pragma unroll
  for (int i = 0; i < 8; i++) { r[i] = a[i]; r[8 + i] = b[i]; }
  return r;
}

__device__ __forceinline__ v16h wmma_a_from_f16(const _Float16* __restrict__ p, int half) {
  h8 a0 = *(const h8*)(p + half * 8);
  h8 a1 = *(const h8*)(p + 16 + half * 8);
  return cat_h8(a0, a1);
}

__device__ __forceinline__ v16h wmma_b_from_f16(const _Float16* __restrict__ p, int half) {
  h8 b0 = *(const h8*)(p + half * 8);
  h8 b1 = *(const h8*)(p + 16 + half * 8);
  return cat_h8(b0, b1);
}

__device__ __forceinline__ v8f wmma32(v16h a, v16h b, v8f c) {
  return __builtin_amdgcn_wmma_f32_16x16x32_f16(false, a, false, b, (short)0, c,
                                                false, false);
}

#if HAVE_TDM
// Issue a TDM load of a 32(keys) x 1024(e) f16 tile (row-major, stride 1024)
// from global into LDS. D# per CDNA5 ISA. 2-D tensor:
// groups 2/3 unused (zero). Tracked on TENSORcnt.
__device__ __forceinline__ void tdm_load_k_tile(const _Float16* gsrc,
                                                unsigned lds_addr) {
  unsigned long long ga = (unsigned long long)(size_t)gsrc;
  u32x4 g0;
  g0[0] = 1u;                                   // count=1 valid descriptor
  g0[1] = lds_addr;                             // LDS byte address
  g0[2] = (unsigned)(ga & 0xFFFFFFFFu);         // global_addr[31:0]
  g0[3] = (unsigned)((ga >> 32) & 0x1FFFFFFu)   // global_addr[56:32]
          | (2u << 30);                         // type=2 ("image")
  i32x8 g1;
  g1[0] = 0x00010000;          // wg_mask=0, data_size=1 (2B), no pad/iter
  g1[1] = (int)(1024u << 16);  // atomic_barrier_addr=0 | tensor_dim0=1024 lo16
  g1[2] = (int)(2048u << 16);  // tensor_dim0 hi=0 | tensor_dim1=2048 lo16
  g1[3] = (int)(1024u << 16);  // tensor_dim1 hi=0 | tile_dim0=1024
  g1[4] = 32;                  // tile_dim1=32 keys, tile_dim2=0
  g1[5] = 1024;                // tensor_dim0_stride[31:0] = 1024 elems
  g1[6] = 0;                   // stride hi | tensor_dim1_stride lo
  g1[7] = 0;
  i32x4 z4 = {0, 0, 0, 0};
#if defined(__clang_major__) && __clang_major__ >= 23
  i32x8 z8 = {0, 0, 0, 0, 0, 0, 0, 0};
  __builtin_amdgcn_tensor_load_to_lds(g0, g1, z4, z4, z8, 0);
#else
  __builtin_amdgcn_tensor_load_to_lds(g0, g1, z4, z4, 0);
#endif
}
#endif

// ---- Kernel 0: one-shot f32 -> f16 conversion (8 elems / thread) -----------
__global__ __launch_bounds__(256) void cvt_f16_kernel(
    const float* __restrict__ in, _Float16* __restrict__ out, int n8) {
  const int i = blockIdx.x * blockDim.x + threadIdx.x;
  if (i < n8) {
    const f4* p = (const f4*)in + (size_t)i * 2;
    f4 c0 = p[0];
    f4 c1 = p[1];
    h8 r;
#pragma unroll
    for (int j = 0; j < 4; j++) {
      r[j]     = (_Float16)c0[j];
      r[4 + j] = (_Float16)c1[j];
    }
    *((h8*)out + i) = r;
  }
}

// ---- Kernel 1: QKV projection GEMM (f16 operands, f32 accum) ---------------
// grid.x = 3 mats * 256 rowTiles * 2 colGroups = 1536, block = 256 (8 waves).
// Each wave: 32(M) x 64(N) tile, K = 1024 in steps of 32. 8 WMMA per K-step.
__global__ __launch_bounds__(256) void qkv_gemm_kernel(
    const _Float16* __restrict__ Xh,
    const _Float16* __restrict__ Whq, const float* __restrict__ bq,
    const _Float16* __restrict__ Whk, const float* __restrict__ bk,
    const _Float16* __restrict__ Whv, const float* __restrict__ bv,
    _Float16* __restrict__ Qh, _Float16* __restrict__ Kh,
    _Float16* __restrict__ Vt) {
  const int tid  = threadIdx.x;
  const int wave = tid >> 5;
  const int lane = tid & 31;
  const int l15  = lane & 15;
  const int half = lane >> 4;

  const int bid      = blockIdx.x;
  const int mat      = bid >> 9;             // 0=Q 1=K 2=V
  const int rem      = bid & 511;
  const int rowTile  = rem >> 1;             // 0..255 over B*S/32
  const int colGroup = rem & 1;
  const int r0       = rowTile * 32;
  const int n0       = colGroup * 512 + wave * 64;

  const _Float16* W = (mat == 0) ? Whq : (mat == 1) ? Whk : Whv;
  const float* bias = (mat == 0) ? bq : (mat == 1) ? bk : bv;

  v8f acc[2][4];
#pragma unroll
  for (int s = 0; s < 2; s++)
#pragma unroll
    for (int t = 0; t < 4; t++)
#pragma unroll
      for (int i = 0; i < 8; i++) acc[s][t][i] = 0.0f;

  const _Float16* arow0 = Xh + (size_t)(r0 + l15) * DD;
  const _Float16* arow1 = Xh + (size_t)(r0 + 16 + l15) * DD;
  for (int k0 = 0; k0 < DD; k0 += 32) {
    if (k0 + 64 < DD) {  // prefetch next-but-one A tile -> global_prefetch_b8
      __builtin_prefetch(arow0 + k0 + 64, 0, 1);
      __builtin_prefetch(arow1 + k0 + 64, 0, 1);
    }
    v16h a0 = wmma_a_from_f16(arow0 + k0, half);
    v16h a1 = wmma_a_from_f16(arow1 + k0, half);
#pragma unroll
    for (int t = 0; t < 4; t++) {
      const _Float16* brow = W + (size_t)(n0 + t * 16 + l15) * DD + k0;
      v16h b = wmma_b_from_f16(brow, half);
      acc[0][t] = wmma32(a0, b, acc[0][t]);
      acc[1][t] = wmma32(a1, b, acc[1][t]);
    }
  }

  const float scale = (mat == 0) ? 0.03125f : 1.0f;  // fold 1/sqrt(D) into Q
#pragma unroll
  for (int s = 0; s < 2; s++) {
#pragma unroll
    for (int t = 0; t < 4; t++) {
      const int n = n0 + t * 16 + l15;
      const float bn = bias[n];
#pragma unroll
      for (int i = 0; i < 8; i++) {
        const int m   = i + half * 8;
        const int row = r0 + s * 16 + m;      // global row in [0, B*S)
        const float v = (acc[s][t][i] + bn) * scale;
        if (mat == 2) {
          const int bidx = row >> 11;         // row / 2048
          const int ss   = row & 2047;
          Vt[(size_t)(bidx * DD + n) * SS + ss] = (_Float16)v;
        } else {
          _Float16* dst = (mat == 0) ? Qh : Kh;
          dst[(size_t)row * DD + n] = (_Float16)v;
        }
      }
    }
  }
}

// ---- Kernel 2: causal flash attention --------------------------------------
// grid.x = B * S/16 = 512, block = 256 (8 waves). Block owns a 16-query tile.
// Wave w owns e-range [w*128, w*128+128) of the output. The 32-key K tile is
// DMA'd into LDS by the Tensor Data Mover (double-buffered, TENSORcnt), so the
// 8 waves read it via ds_load instead of 8x-redundant global loads.
__global__ __launch_bounds__(256) void flash_attn_kernel(
    const _Float16* __restrict__ Qh, const _Float16* __restrict__ Kh,
    const _Float16* __restrict__ Vt, float* __restrict__ out) {
#if HAVE_TDM
  __shared__ __align__(16) _Float16 sK[2][32 * DD];  // 2 x 64KB K tiles
#endif
  __shared__ __align__(16) float    sS[16 * 32];     // f32 score tile 16x32
  __shared__ __align__(16) _Float16 sP[16 * 32];     // f16 probabilities
  __shared__ float sAlpha[16];
  __shared__ float sL[16];

  const int tid  = threadIdx.x;
  const int wave = tid >> 5;
  const int lane = tid & 31;
  const int l15  = lane & 15;
  const int half = lane >> 4;

  const int b  = blockIdx.x >> 7;    // S/16 = 128 query tiles per batch
  const int qt = blockIdx.x & 127;
  const int q0 = qt * 16;
  const size_t rowBase = (size_t)b * SS;

  // Preload this wave's Q slice: 16 rows x 128 e (4 K-chunks of 32)
  v16h qv[4];
  {
    const _Float16* qrow = Qh + (rowBase + q0 + l15) * DD;
#pragma unroll
    for (int c = 0; c < 4; c++)
      qv[c] = wmma_a_from_f16(qrow + wave * 128 + c * 32, half);
  }

  v8f acc[8];
#pragma unroll
  for (int c = 0; c < 8; c++)
#pragma unroll
    for (int i = 0; i < 8; i++) acc[c][i] = 0.0f;

  float mrun = -1e30f, lrun = 0.0f;  // meaningful only for tid < 16

  const int numIter = (q0 + 15) / 32 + 1;  // causal bound, 32 keys per iter

#if HAVE_TDM
  if (wave == 0)  // TDM ignores EXEC: one instruction issue per wave
    tdm_load_k_tile(Kh + rowBase * DD, (unsigned)(size_t)(void*)&sK[0][0]);
#endif

  for (int j = 0; j < numIter; j++) {
    const int kBase = j * 32;

#if HAVE_TDM
    if (wave == 0) {
      if ((j + 1) < numIter) {  // prefetch next K tile into the other buffer
        tdm_load_k_tile(Kh + (rowBase + kBase + 32) * DD,
                        (unsigned)(size_t)(void*)&sK[(j + 1) & 1][0]);
        __builtin_amdgcn_s_wait_tensorcnt(1);  // wait for the older transfer
      } else {
        __builtin_amdgcn_s_wait_tensorcnt(0);  // last tile: drain all
      }
    }
#endif
    sS[tid] = 0.0f;
    sS[tid + 256] = 0.0f;
    __syncthreads();

    // Partial scores over this wave's 128 e's: 2 key subtiles x 4 K-chunks
    v8f ps[2];
#pragma unroll
    for (int t = 0; t < 2; t++) {
#pragma unroll
      for (int i = 0; i < 8; i++) ps[t][i] = 0.0f;
#if HAVE_TDM
      const _Float16* krow = &sK[j & 1][(t * 16 + l15) * DD];  // LDS (ds_load)
#else
      const _Float16* krow = Kh + (rowBase + kBase + t * 16 + l15) * DD;
#endif
#pragma unroll
      for (int c = 0; c < 4; c++) {
        v16h kb = wmma_b_from_f16(krow + wave * 128 + c * 32, half);
        ps[t] = wmma32(qv[c], kb, ps[t]);
      }
    }
#pragma unroll
    for (int t = 0; t < 2; t++)
#pragma unroll
      for (int i = 0; i < 8; i++)
        atomicAdd(&sS[(i + half * 8) * 32 + t * 16 + l15], ps[t][i]);
    __syncthreads();

    // Online softmax for the 16x32 tile (wave 0, lanes 0..15; one row each)
    if (tid < 16) {
      const int m  = tid;
      const int qg = q0 + m;
      float mx = -1e30f;
      float sv[32];
#pragma unroll
      for (int k = 0; k < 32; k++) {
        float s = sS[m * 32 + k];
        if (kBase + k > qg) s = -1e30f;  // causal mask
        sv[k] = s;
        mx = fmaxf(mx, s);
      }
      const float mnew  = fmaxf(mrun, mx);
      const float alpha = __expf(mrun - mnew);
      float sum = 0.0f;
#pragma unroll
      for (int k = 0; k < 32; k++) {
        const float p = __expf(sv[k] - mnew);
        sum += p;
        sP[m * 32 + k] = (_Float16)p;
      }
      lrun = lrun * alpha + sum;
      mrun = mnew;
      sAlpha[m] = alpha;
    }
    __syncthreads();

    // Rescale O accumulators and add P @ V over this wave's 128 e's
    float al[8];
#pragma unroll
    for (int i = 0; i < 8; i++) al[i] = sAlpha[i + half * 8];
    v16h pa = wmma_a_from_f16(sP + l15 * 32, half);
#pragma unroll
    for (int c = 0; c < 8; c++) {
#pragma unroll
      for (int i = 0; i < 8; i++) acc[c][i] *= al[i];
      const int e = wave * 128 + c * 16 + l15;
      const _Float16* vrow = Vt + (size_t)(b * DD + e) * SS + kBase;
      v16h vb = wmma_b_from_f16(vrow, half);
      acc[c] = wmma32(pa, vb, acc[c]);
    }
  }

  if (tid < 16) sL[tid] = lrun;
  __syncthreads();

  float li[8];
#pragma unroll
  for (int i = 0; i < 8; i++) li[i] = 1.0f / sL[i + half * 8];
#pragma unroll
  for (int c = 0; c < 8; c++) {
    const int e = wave * 128 + c * 16 + l15;
#pragma unroll
    for (int i = 0; i < 8; i++) {
      const int m = i + half * 8;
      out[(rowBase + q0 + m) * DD + e] = acc[c][i] * li[i];
    }
  }
}

// ---- Host launch ------------------------------------------------------------
extern "C" void kernel_launch(void* const* d_in, const int* in_sizes, int n_in,
                              void* d_out, int out_size, void* d_ws,
                              size_t ws_size, hipStream_t stream) {
  const float* x  = (const float*)d_in[0];
  const float* Wq = (const float*)d_in[1];
  const float* bq = (const float*)d_in[2];
  const float* Wk = (const float*)d_in[3];
  const float* bk = (const float*)d_in[4];
  const float* Wv = (const float*)d_in[5];
  const float* bv = (const float*)d_in[6];

  const size_t mat_elems = (size_t)BB * SS * DD;  // 8M halves = 16 MB each
  const size_t w_elems   = (size_t)DD * DD;       // 1M halves = 2 MB each
  _Float16* Qh  = (_Float16*)d_ws;
  _Float16* Kh  = Qh + mat_elems;
  _Float16* Vt  = Kh + mat_elems;   // stored as [B][D][S]
  _Float16* Xh  = Vt + mat_elems;
  _Float16* Whq = Xh + mat_elems;
  _Float16* Whk = Whq + w_elems;
  _Float16* Whv = Whk + w_elems;

  // One-shot f32 -> f16 conversions (hoists all cvt out of the GEMM loop).
  cvt_f16_kernel<<<dim3((unsigned)(mat_elems / 8 / 256)), dim3(256), 0, stream>>>(
      x, Xh, (int)(mat_elems / 8));
  cvt_f16_kernel<<<dim3((unsigned)(w_elems / 8 / 256)), dim3(256), 0, stream>>>(
      Wq, Whq, (int)(w_elems / 8));
  cvt_f16_kernel<<<dim3((unsigned)(w_elems / 8 / 256)), dim3(256), 0, stream>>>(
      Wk, Whk, (int)(w_elems / 8));
  cvt_f16_kernel<<<dim3((unsigned)(w_elems / 8 / 256)), dim3(256), 0, stream>>>(
      Wv, Whv, (int)(w_elems / 8));

  qkv_gemm_kernel<<<dim3(1536), dim3(256), 0, stream>>>(
      Xh, Whq, bq, Whk, bk, Whv, bv, Qh, Kh, Vt);
  flash_attn_kernel<<<dim3(512), dim3(256), 0, stream>>>(
      Qh, Kh, Vt, (float*)d_out);
}
